// MultiHeadAttention_62096637165767
// MI455X (gfx1250) — matrix operand order repaired and checked by reading, not compiled, not run on hardware
//
#include <hip/hip_runtime.h>

typedef __bf16 bf16;
typedef __attribute__((ext_vector_type(16))) __bf16 v16bf;
typedef __attribute__((ext_vector_type(8)))  __bf16 v8bf;
typedef __attribute__((ext_vector_type(4)))  __bf16 v4bf;
typedef __attribute__((ext_vector_type(8)))  float  v8f;

#define DEV static __device__ __forceinline__

// ---------------------------------------------------------------------------
// WMMA fragment loaders (contiguous k order used by this code for v_wmma_f32_16x16x32_bf16)
// A (16x32, MxK): lane = sel*16 + m ; halfs 0..15 = K[sel*16 .. +15]
//                 of row m
// B (32x16, KxN) from Bt stored [N][K]: lane = sel*16 + n ;
//                 halfs 0..15 = K[sel*16 .. +15] of column n
// ---------------------------------------------------------------------------

DEV v16bf load_a_f32(const float* __restrict__ base, long long lda, int lane) {
  const int m = lane & 15, sel = lane >> 4;
  const float* p = base + (long long)m * lda + sel * 16;
  v8f lo = *(const v8f*)(p);
  v8f hi = *(const v8f*)(p + 8);
  v16bf r;
#pragma unroll
  for (int i = 0; i < 8; ++i) { r[i] = (bf16)lo[i]; r[i + 8] = (bf16)hi[i]; }
  return r;
}

DEV v16bf load_b_frag(const bf16* __restrict__ base, long long ldb, int lane) {
  const int n = lane & 15, sel = lane >> 4;
  return *(const v16bf*)(base + (long long)n * ldb + sel * 16);
}

DEV v16bf combine16(v8bf lo, v8bf hi) {
  v16bf r;
#pragma unroll
  for (int i = 0; i < 8; ++i) { r[i] = lo[i]; r[i + 8] = hi[i]; }
  return r;
}

// Issue one async global->LDS 16-byte copy (tracked by ASYNCcnt).
DEV void async_copy_b128(unsigned lds_off, const void* gaddr) {
  asm volatile("global_load_async_to_lds_b128 %0, %1, off"
               :: "v"(lds_off), "v"(gaddr)
               : "memory");
}

DEV void wait_async0() {
  asm volatile("s_wait_asynccnt 0x0" ::: "memory");
}

// ---------------------------------------------------------------------------
// Generic batched NT GEMM:  C[m,n] = alpha * sum_k A[m,k] * Bt[n,k]
// Batch index z decomposed as (b1 = z/HH, b2 = z%HH) with independent strides.
// Wave tile 64x64 (4x4 wmma frags); block = 8 waves arranged WGM x WGN.
// bf16-A path: double-buffered async global->LDS staging of the block's
//   A (WGM*64 x 32) and B (WGN*64 x 32) k-slices, fragments gathered from LDS.
// f32-A path (attn @ V): direct global loads with in-register f32->bf16.
// ---------------------------------------------------------------------------
template <typename AT, typename CT, int WGM, int WGN>
__global__ __launch_bounds__(256) void gemm_nt_wmma(
    const AT* __restrict__ A, long long lda, long long sA1, long long sA2,
    const bf16* __restrict__ Bt, long long ldb, long long sB1, long long sB2,
    CT* __restrict__ C, long long ldc, long long sC1, long long sC2,
    int K, int HH, float alpha)
{
  const int tid  = threadIdx.x;
  const int lane = tid & 31;
  const int wid  = tid >> 5;
  const int wm = wid / WGN, wn = wid % WGN;
  const long long m0 = (long long)blockIdx.y * (WGM * 64) + wm * 64;
  const long long n0 = (long long)blockIdx.x * (WGN * 64) + wn * 64;

  const int bz = blockIdx.z;
  const int b1 = bz / HH, b2 = bz % HH;
  const AT*   a_base = A  + (long long)b1 * sA1 + (long long)b2 * sA2;
  const bf16* b_base = Bt + (long long)b1 * sB1 + (long long)b2 * sB2;
  CT*         c_base = C  + (long long)b1 * sC1 + (long long)b2 * sC2;

  v8f acc[4][4] = {};

  if constexpr (sizeof(AT) == 2) {
    // ---------------- LDS-staged bf16 path ----------------
    constexpr int RA = WGM * 64;              // A rows per block tile
    constexpr int RB = WGN * 64;              // B rows per block tile
    __shared__ bf16 shA[2][RA * 32];
    __shared__ bf16 shB[2][RB * 32];

    const long long mBase = (long long)blockIdx.y * RA;
    const long long nBase = (long long)blockIdx.x * RB;
    const bf16* Ab = (const bf16*)a_base;

    // LDS aperture keeps the byte offset in the low 32 bits.
    const unsigned ldsA = (unsigned)(size_t)(&shA[0][0]);
    const unsigned ldsB = (unsigned)(size_t)(&shB[0][0]);

    constexpr int CA = RA * 4 / 256;          // 16B chunks per thread (A)
    constexpr int CB = RB * 4 / 256;          // 16B chunks per thread (B)

    auto stage = [&](int buf, int kk) {
#pragma unroll
      for (int c = 0; c < CA; ++c) {
        const int lin = tid * CA + c;
        const int row = lin >> 2, ch = lin & 3;   // 4 chunks per 32-elem row
        const bf16* gp = Ab + (mBase + row) * lda + kk + ch * 8;
        const unsigned lo = ldsA + (unsigned)(buf * RA * 32 + row * 32 + ch * 8) * 2u;
        async_copy_b128(lo, gp);
      }
#pragma unroll
      for (int c = 0; c < CB; ++c) {
        const int lin = tid * CB + c;
        const int row = lin >> 2, ch = lin & 3;
        const bf16* gp = b_base + (nBase + row) * ldb + kk + ch * 8;
        const unsigned lo = ldsB + (unsigned)(buf * RB * 32 + row * 32 + ch * 8) * 2u;
        async_copy_b128(lo, gp);
      }
    };

    stage(0, 0);
    int buf = 0;
    for (int kk = 0; kk < K; kk += 32) {
      wait_async0();
      __syncthreads();                         // staged tile visible to block
      if (kk + 32 < K) stage(buf ^ 1, kk + 32);

      const int ml = lane & 15, sel = lane >> 4;
      v16bf af[4], bfr[4];
#pragma unroll
      for (int i = 0; i < 4; ++i) {
        const bf16* pa = &shA[buf][(wm * 64 + i * 16 + ml) * 32 + sel * 16];
        af[i] = combine16(*(const v8bf*)pa, *(const v8bf*)(pa + 8));
        bfr[i] = *(const v16bf*)&shB[buf][(wn * 64 + i * 16 + ml) * 32 + sel * 16];
      }
#pragma unroll
      for (int i = 0; i < 4; ++i)
#pragma unroll
        for (int j = 0; j < 4; ++j)
          acc[i][j] = __builtin_amdgcn_wmma_f32_16x16x32_bf16(
              false, af[i], false, bfr[j], (short)0, acc[i][j], false, false);

      __syncthreads();                         // all reads done before overwrite
      buf ^= 1;
    }
  } else {
    // ---------------- direct-load f32-A path ----------------
    for (int kk = 0; kk < K; kk += 32) {
      v16bf af[4], bfr[4];
#pragma unroll
      for (int i = 0; i < 4; ++i) {
        af[i]  = load_a_f32((const float*)a_base + (m0 + i * 16) * lda + kk, lda, lane);
        bfr[i] = load_b_frag(b_base + (n0 + i * 16) * ldb + kk, ldb, lane);
      }
#pragma unroll
      for (int i = 0; i < 4; ++i)
#pragma unroll
        for (int j = 0; j < 4; ++j)
          acc[i][j] = __builtin_amdgcn_wmma_f32_16x16x32_bf16(
              false, af[i], false, bfr[j], (short)0, acc[i][j], false, false);
    }
  }

  const int sel = lane >> 4, nl = lane & 15;
#pragma unroll
  for (int i = 0; i < 4; ++i)
#pragma unroll
    for (int j = 0; j < 4; ++j) {
      CT* cp = c_base + (m0 + i * 16 + sel * 8) * ldc + n0 + j * 16 + nl;
#pragma unroll
      for (int r = 0; r < 8; ++r) {
        cp[(long long)r * ldc] = (CT)(acc[i][j][r] * alpha);
      }
    }
}

// ---------------------------------------------------------------------------
// f32 -> bf16 conversion (vectorized by 4)
// ---------------------------------------------------------------------------
__global__ __launch_bounds__(256) void cvt_f32_bf16(
    const float* __restrict__ in, bf16* __restrict__ out, int n4)
{
  int idx = blockIdx.x * 256 + threadIdx.x;
  if (idx < n4) {
    float4 v = ((const float4*)in)[idx];
    v4bf o;
    o[0] = (bf16)v.x; o[1] = (bf16)v.y; o[2] = (bf16)v.z; o[3] = (bf16)v.w;
    ((v4bf*)out)[idx] = o;
  }
}

// ---------------------------------------------------------------------------
// Vt[(b*H+h)][d][s] = Vproj[b][s][h*DK+d]   (bf16, per-head transpose)
// ---------------------------------------------------------------------------
__global__ __launch_bounds__(256) void transpose_v(
    const bf16* __restrict__ Vp, bf16* __restrict__ Vt)
{
  const int S = 2048, D = 1024, H = 16, DK = 64;
  long long idx = (long long)blockIdx.x * 256 + threadIdx.x;
  int s = (int)(idx % S);
  long long t = idx / S;
  int d = (int)(t % DK);
  long long bh = t / DK;
  int h = (int)(bh % H);
  int b = (int)(bh / H);
  Vt[idx] = Vp[(long long)b * S * D + (long long)s * D + h * DK + d];
}

// ---------------------------------------------------------------------------
// In-place row softmax over S=2048 (one block per row, wave32 reductions)
// ---------------------------------------------------------------------------
__global__ __launch_bounds__(256) void softmax_rows(float* __restrict__ attn)
{
  const int S = 2048;
  float* row = attn + (long long)blockIdx.x * S;
  const int t = threadIdx.x;

  __shared__ float sm[8];
  __shared__ float ss[8];

  float x[8];
  float m = -3.4e38f;
#pragma unroll
  for (int i = 0; i < 8; ++i) { x[i] = row[t + i * 256]; m = fmaxf(m, x[i]); }
#pragma unroll
  for (int off = 16; off >= 1; off >>= 1) m = fmaxf(m, __shfl_xor(m, off, 32));
  if ((t & 31) == 0) sm[t >> 5] = m;
  __syncthreads();
  float bm = sm[0];
#pragma unroll
  for (int i = 1; i < 8; ++i) bm = fmaxf(bm, sm[i]);

  float s = 0.f;
#pragma unroll
  for (int i = 0; i < 8; ++i) { x[i] = __expf(x[i] - bm); s += x[i]; }
#pragma unroll
  for (int off = 16; off >= 1; off >>= 1) s += __shfl_xor(s, off, 32);
  if ((t & 31) == 0) ss[t >> 5] = s;
  __syncthreads();
  float bs = 0.f;
#pragma unroll
  for (int i = 0; i < 8; ++i) bs += ss[i];
  float inv = 1.0f / bs;
#pragma unroll
  for (int i = 0; i < 8; ++i) row[t + i * 256] = x[i] * inv;
}

// ---------------------------------------------------------------------------
// Host-side orchestration
// ---------------------------------------------------------------------------
extern "C" void kernel_launch(void* const* d_in, const int* in_sizes, int n_in,
                              void* d_out, int out_size, void* d_ws, size_t ws_size,
                              hipStream_t stream) {
  (void)in_sizes; (void)n_in; (void)out_size; (void)ws_size;
  const int B = 2, S = 2048, D = 1024, H = 16, DK = 64;
  const long long MS = (long long)B * S;          // 4096 token rows
  const long long ND = D;

  const float* q_f  = (const float*)d_in[0];
  const float* k_f  = (const float*)d_in[1];
  const float* v_f  = (const float*)d_in[2];
  const float* wq_f = (const float*)d_in[3];
  const float* wk_f = (const float*)d_in[4];
  const float* wv_f = (const float*)d_in[5];
  const float* wo_f = (const float*)d_in[6];

  char* ws = (char*)d_ws;
  size_t off = 0;
  auto carve = [&](size_t bytes) -> char* {
    char* p = ws + off;
    off = (off + bytes + 255) & ~(size_t)255;
    return p;
  };

  bf16* qb  = (bf16*)carve(MS * D * sizeof(bf16));
  bf16* kb  = (bf16*)carve(MS * D * sizeof(bf16));
  bf16* vb  = (bf16*)carve(MS * D * sizeof(bf16));
  bf16* wqb = (bf16*)carve((size_t)D * D * sizeof(bf16));
  bf16* wkb = (bf16*)carve((size_t)D * D * sizeof(bf16));
  bf16* wvb = (bf16*)carve((size_t)D * D * sizeof(bf16));
  bf16* wob = (bf16*)carve((size_t)D * D * sizeof(bf16));
  bf16* Qp  = (bf16*)carve(MS * D * sizeof(bf16));
  bf16* Kp  = (bf16*)carve(MS * D * sizeof(bf16));
  bf16* Vp  = (bf16*)carve(MS * D * sizeof(bf16));
  // Reuse dead buffers: qb is dead after Q projection, kb after K projection.
  bf16* Vt   = qb;   // [B*H][DK][S]
  bf16* ctxb = kb;   // [B*S][D]

  float* out  = (float*)d_out;
  float* attn = out + (long long)B * S * D;       // [B,H,S,S] f32

  dim3 blk(256);

  // 1) f32 -> bf16 copies
  {
    int nt = (int)(MS * D / 4);
    int nw = (int)((long long)D * D / 4);
    cvt_f32_bf16<<<(nt + 255) / 256, blk, 0, stream>>>(q_f,  qb,  nt);
    cvt_f32_bf16<<<(nt + 255) / 256, blk, 0, stream>>>(k_f,  kb,  nt);
    cvt_f32_bf16<<<(nt + 255) / 256, blk, 0, stream>>>(v_f,  vb,  nt);
    cvt_f32_bf16<<<(nw + 255) / 256, blk, 0, stream>>>(wq_f, wqb, nw);
    cvt_f32_bf16<<<(nw + 255) / 256, blk, 0, stream>>>(wk_f, wkb, nw);
    cvt_f32_bf16<<<(nw + 255) / 256, blk, 0, stream>>>(wv_f, wvb, nw);
    cvt_f32_bf16<<<(nw + 255) / 256, blk, 0, stream>>>(wo_f, wob, nw);
  }

  // 2) Q/K/V projections: [4096,1024] x [1024,1024]^T  (NT form, LDS-staged)
  {
    dim3 g((unsigned)(ND / 128), (unsigned)(MS / 256), 1);
    gemm_nt_wmma<bf16, bf16, 4, 2><<<g, blk, 0, stream>>>(
        qb, D, 0, 0, wqb, D, 0, 0, Qp, D, 0, 0, D, 1, 1.0f);
    gemm_nt_wmma<bf16, bf16, 4, 2><<<g, blk, 0, stream>>>(
        kb, D, 0, 0, wkb, D, 0, 0, Kp, D, 0, 0, D, 1, 1.0f);
    gemm_nt_wmma<bf16, bf16, 4, 2><<<g, blk, 0, stream>>>(
        vb, D, 0, 0, wvb, D, 0, 0, Vp, D, 0, 0, D, 1, 1.0f);
  }

  // 3) scores = Q_h K_h^T / sqrt(DK), per (b,h), f32 into d_out attn region
  {
    dim3 g((unsigned)(S / 128), (unsigned)(S / 256), (unsigned)(B * H));
    gemm_nt_wmma<bf16, float, 4, 2><<<g, blk, 0, stream>>>(
        Qp, D, (long long)S * D, DK,
        Kp, D, (long long)S * D, DK,
        attn, S, (long long)H * S * S, (long long)S * S,
        DK, H, 0.125f);
  }

  // 4) softmax over last dim, in place
  softmax_rows<<<B * H * S, blk, 0, stream>>>(attn);

  // 5) per-head transpose of V for K-contiguous B operand
  transpose_v<<<(unsigned)((long long)B * H * DK * S / 256), blk, 0, stream>>>(Vp, Vt);

  // 6) ctx = attn @ V_h  (A = f32 attn converted in-register, NT vs Vt)
  {
    dim3 g(1, (unsigned)(S / 512), (unsigned)(B * H));
    gemm_nt_wmma<float, bf16, 8, 1><<<g, blk, 0, stream>>>(
        attn, S, (long long)H * S * S, (long long)S * S,
        Vt,   S, (long long)H * DK * S, (long long)DK * S,
        ctxb, D, (long long)S * D, DK,
        S, H, 1.0f);
  }

  // 7) out = ctx @ Wo^T  -> f32 into d_out
  {
    dim3 g((unsigned)(ND / 128), (unsigned)(MS / 256), 1);
    gemm_nt_wmma<bf16, float, 4, 2><<<g, blk, 0, stream>>>(
        ctxb, D, 0, 0, wob, D, 0, 0, out, D, 0, 0, D, 1, 1.0f);
  }
}
